// GCNEncoder_71124658421873
// MI455X (gfx1250) — matrix operand order repaired and checked by reading, not compiled, not run on hardware
//
#include <hip/hip_runtime.h>
#include <math.h>

// ---------------------------------------------------------------------------
// 2-layer GCN (PyG GCNConv semantics) for MI455X / gfx1250.
//   deg[v] = indeg(v)+1 ; dis = rsqrt(deg)
//   layer: t = A @ W (WMMA f16 in / f32 acc) ;
//          out = sum_{(s,d)} t[s]*dis[s]*dis[d] + t[i]*dis[i]^2 + bias ;
//          ReLU between layers (fused into GEMM2's A load).
// GEMM: persistent waves, B (weights) converted f32->f16 once and held in
// registers (2 col tiles x 4 ktiles = 64 VGPRs), K fully unrolled -> 8 WMMAs
// per row tile per wave, A loaded as 4x global_load_b128 per ktile.
// Scatter: 205M (layer 1) / 102M (layer 2) hardware global_atomic_add_f32 (forced via inline
// asm so no CAS-loop fallback); working set ~130MB is L2-resident (192MB).
// ---------------------------------------------------------------------------

typedef __attribute__((ext_vector_type(16))) _Float16 v16h;
typedef __attribute__((ext_vector_type(8)))  float    v8f;

// Non-returning hardware f32 atomic add at device scope (L2-side RMW).
// Uses STOREcnt only; s_endpgm's implicit wait-idle covers completion.
__device__ __forceinline__ void atomicAddF32(float* p, float v) {
  asm volatile("global_atomic_add_f32 %0, %1, off scope:SCOPE_DEV"
               :
               : "v"(p), "v"(v)
               : "memory");
}

__global__ void k_fill1(float* __restrict__ p, int n) {
  int i = blockIdx.x * blockDim.x + threadIdx.x;
  if (i < n) p[i] = 1.0f;
}

__global__ void k_deg(const int* __restrict__ dst, float* __restrict__ deg, int E) {
  int e = blockIdx.x * blockDim.x + threadIdx.x;
  if (e < E) atomicAddF32(&deg[dst[e]], 1.0f);
}

__global__ void k_rsqrt(const float* __restrict__ deg, float* __restrict__ dis, int n) {
  int i = blockIdx.x * blockDim.x + threadIdx.x;
  if (i < n) {
    float d = deg[i];
    dis[i] = d > 0.f ? rsqrtf(d) : 0.f;
  }
}

// out[i,c] = t[i,c]*dis[i]^2 + bias[c]  (self-loop message + bias). grid=n, block=M
__global__ void k_init(const float* __restrict__ t, const float* __restrict__ dis,
                       const float* __restrict__ bias, float* __restrict__ out, int M) {
  int i = blockIdx.x;
  int c = threadIdx.x;
  float d = dis[i];
  unsigned idx = (unsigned)i * M + c;
  out[idx] = t[idx] * d * d + bias[c];
}

// One thread handles 4 consecutive columns of one edge's message.
// threads-per-edge = M/4 = 1<<shift, so no integer division anywhere.
__global__ void k_scatter(const int* __restrict__ src, const int* __restrict__ dst,
                          const float* __restrict__ dis, const float* __restrict__ t,
                          float* __restrict__ out, int E, int M, int shift) {
  int tid = threadIdx.x;
  int e = blockIdx.x * (blockDim.x >> shift) + (tid >> shift);
  if (e >= E) return;
  int c4 = (tid & ((1 << shift) - 1)) << 2;
  int s = src[e], d = dst[e];
  float norm = dis[s] * dis[d];
  const float4 v = *(const float4*)(t + (unsigned)s * M + c4);
  float* o = out + (unsigned)d * M + c4;
  atomicAddF32(o + 0, v.x * norm);
  atomicAddF32(o + 1, v.y * norm);
  atomicAddF32(o + 2, v.z * norm);
  atomicAddF32(o + 3, v.w * norm);
}

// Cout[n,M] = f(Asrc)[n,KT] @ W[KT,M], f16 WMMA with f32 accumulate.
// MODE 0: A row gathered from embedding table via types[] (fused emb lookup).
// MODE 1: A = relu(Asrc) fused into the f32->f16 convert.
// Persistent waves: each wave owns COLS column tiles, preloads all B fragments
// into registers, then strides over row tiles.
template <int MODE, int KT, int COLS>
__global__ void gcn_gemm_wmma(const float* __restrict__ Asrc,
                              const int* __restrict__ types,
                              const float* __restrict__ W,
                              float* __restrict__ Cout,
                              int n, int M, int rowWaves) {
  constexpr int KTILES = KT / 32;
  int wave = (int)((blockIdx.x * blockDim.x + threadIdx.x) >> 5);
  int lane = threadIdx.x & 31;
  int colGroups = M / (16 * COLS);
  int colGrp = wave % colGroups;
  int rt0 = wave / colGroups;            // starting row tile for this wave
  int rowTiles = (n + 15) >> 4;
  int half = lane >> 4;                  // 0: lanes 0-15, 1: lanes 16-31
  int r = lane & 15;

  // ---- preload B fragments once (16-bit B layout matching A: lane holds
  // column n0+r; VGPR j holds K = half*8 + (j>>2)*8 + 2j{,+1}). W is f32, L2-resident.
  v16h bf[COLS][KTILES];
#pragma unroll
  for (int c = 0; c < COLS; ++c) {
    int n0 = ((colGrp * COLS + c) << 4) + r;
#pragma unroll
    for (int kt = 0; kt < KTILES; ++kt) {
#pragma unroll
      for (int j = 0; j < 8; ++j) {
        int k = kt * 32 + (half << 3) + ((j >> 2) << 3) + (j << 1);
        bf[c][kt][2 * j]     = (_Float16)W[k * M + n0];
        bf[c][kt][2 * j + 1] = (_Float16)W[(k + 1) * M + n0];
      }
    }
  }

  for (int rt = rt0; rt < rowTiles; rt += rowWaves) {
    int m0 = rt << 4;
    int row = m0 + r;
    if (row > n - 1) row = n - 1;        // clamp loads: EXEC stays all-1s
    const float* arow = (MODE == 0) ? Asrc + (unsigned)types[row] * KT
                                    : Asrc + (unsigned)row * KT;
    v8f acc[COLS] = {};
#pragma unroll
    for (int kt = 0; kt < KTILES; ++kt) {
      // A layout: lane holds row m0+r; halves 0..7 are K = kt*32+half*8+0..7,
      // halves 8..15 are K = kt*32+16+half*8+0..7 -> two contiguous 32B runs.
      const float4* p0 = (const float4*)(arow + kt * 32 + (half << 3));
      const float4* p1 = (const float4*)(arow + kt * 32 + 16 + (half << 3));
      float4 q0 = p0[0], q1 = p0[1], q2 = p1[0], q3 = p1[1];
      if (MODE == 1) {
        q0.x = fmaxf(q0.x, 0.f); q0.y = fmaxf(q0.y, 0.f); q0.z = fmaxf(q0.z, 0.f); q0.w = fmaxf(q0.w, 0.f);
        q1.x = fmaxf(q1.x, 0.f); q1.y = fmaxf(q1.y, 0.f); q1.z = fmaxf(q1.z, 0.f); q1.w = fmaxf(q1.w, 0.f);
        q2.x = fmaxf(q2.x, 0.f); q2.y = fmaxf(q2.y, 0.f); q2.z = fmaxf(q2.z, 0.f); q2.w = fmaxf(q2.w, 0.f);
        q3.x = fmaxf(q3.x, 0.f); q3.y = fmaxf(q3.y, 0.f); q3.z = fmaxf(q3.z, 0.f); q3.w = fmaxf(q3.w, 0.f);
      }
      v16h a;
      a[0]  = (_Float16)q0.x; a[1]  = (_Float16)q0.y; a[2]  = (_Float16)q0.z; a[3]  = (_Float16)q0.w;
      a[4]  = (_Float16)q1.x; a[5]  = (_Float16)q1.y; a[6]  = (_Float16)q1.z; a[7]  = (_Float16)q1.w;
      a[8]  = (_Float16)q2.x; a[9]  = (_Float16)q2.y; a[10] = (_Float16)q2.z; a[11] = (_Float16)q2.w;
      a[12] = (_Float16)q3.x; a[13] = (_Float16)q3.y; a[14] = (_Float16)q3.z; a[15] = (_Float16)q3.w;
#pragma unroll
      for (int c = 0; c < COLS; ++c)
        acc[c] = __builtin_amdgcn_wmma_f32_16x16x32_f16(
            false, a, false, bf[c][kt], (short)0, acc[c], false, false);
    }
    // D layout: VGPR v, lane -> row m0 + v + 8*half, col n0 + r.
    // 32-bit offsets; wave-uniform full/tail branch keeps hot path clean.
    if (m0 + 16 <= n) {
#pragma unroll
      for (int c = 0; c < COLS; ++c) {
        unsigned base = (unsigned)(m0 + (half << 3)) * M + ((colGrp * COLS + c) << 4) + r;
#pragma unroll
        for (int v = 0; v < 8; ++v) Cout[base + (unsigned)(v * M)] = acc[c][v];
      }
    } else {
#pragma unroll
      for (int c = 0; c < COLS; ++c) {
        unsigned base = (unsigned)(m0 + (half << 3)) * M + ((colGrp * COLS + c) << 4) + r;
#pragma unroll
        for (int v = 0; v < 8; ++v)
          if (m0 + (half << 3) + v < n) Cout[base + (unsigned)(v * M)] = acc[c][v];
      }
    }
  }
}

static inline size_t alignup(size_t x) { return (x + 255) & ~(size_t)255; }

extern "C" void kernel_launch(void* const* d_in, const int* in_sizes, int n_in,
                              void* d_out, int out_size, void* d_ws, size_t ws_size,
                              hipStream_t stream) {
  (void)n_in; (void)out_size; (void)ws_size;
  const int n    = in_sizes[0];        // 100000 nodes
  const int E    = in_sizes[1] / 2;    // 1.6M edges
  const int hid  = in_sizes[4];        // 128
  const int outd = in_sizes[6];        // 64

  const int*   types = (const int*)d_in[0];
  const int*   src   = (const int*)d_in[1];
  const int*   dst   = src + E;
  const float* emb   = (const float*)d_in[2];
  const float* W1    = (const float*)d_in[3];
  const float* b1    = (const float*)d_in[4];
  const float* W2    = (const float*)d_in[5];
  const float* b2    = (const float*)d_in[6];
  float* out = (float*)d_out;

  // workspace: deg, dis, t1[n,hid], a1[n,hid], t2[n,outd]  (~130MB)
  char* ws = (char*)d_ws;
  size_t off = 0;
  float* deg = (float*)(ws + off); off = alignup(off + (size_t)n * 4);
  float* dis = (float*)(ws + off); off = alignup(off + (size_t)n * 4);
  float* t1  = (float*)(ws + off); off = alignup(off + (size_t)n * hid * 4);
  float* a1  = (float*)(ws + off); off = alignup(off + (size_t)n * hid * 4);
  float* t2  = (float*)(ws + off); off = alignup(off + (size_t)n * outd * 4);

  const int B = 256;
  const int ROWWAVES = 1024;           // persistent waves per column group

  // degrees (self-loop contributes 1) and normalization
  k_fill1<<<dim3((n + B - 1) / B), dim3(B), 0, stream>>>(deg, n);
  k_deg<<<dim3((E + B - 1) / B), dim3(B), 0, stream>>>(dst, deg, E);
  k_rsqrt<<<dim3((n + B - 1) / B), dim3(B), 0, stream>>>(deg, dis, n);

  // layer 1: t1 = emb[types] @ W1
  {
    int colGroups = hid / 32;                       // COLS=2
    int waves = colGroups * ROWWAVES;               // multiple of 8
    gcn_gemm_wmma<0, 128, 2><<<dim3(waves / 8), dim3(B), 0, stream>>>(
        emb, types, W1, t1, n, hid, ROWWAVES);
  }
  k_init<<<dim3(n), dim3(hid), 0, stream>>>(t1, dis, b1, a1, hid);
  {
    int epb = B >> 5;                               // hid/4 = 32 threads/edge
    k_scatter<<<dim3((E + epb - 1) / epb), dim3(B), 0, stream>>>(
        src, dst, dis, t1, a1, E, hid, 5);
  }

  // layer 2: t2 = relu(a1) @ W2
  {
    int colGroups = outd / 32;
    int waves = colGroups * ROWWAVES;
    gcn_gemm_wmma<1, 128, 2><<<dim3(waves / 8), dim3(B), 0, stream>>>(
        a1, nullptr, W2, t2, n, outd, ROWWAVES);
  }
  k_init<<<dim3(n), dim3(outd), 0, stream>>>(t2, dis, b2, out, outd);
  {
    int epb = B >> 4;                               // outd/4 = 16 threads/edge
    k_scatter<<<dim3((E + epb - 1) / epb), dim3(B), 0, stream>>>(
        src, dst, dis, t2, out, E, outd, 4);
  }
}
